// HolographicAttention_34213709480634
// MI455X (gfx1250) — matrix operand order repaired and checked by reading, not compiled, not run on hardware
//
#include <hip/hip_runtime.h>
#include <hip/hip_bf16.h>

#define D_MODEL 1024
#define N_HEADS 16
#define HEAD_DIM 64
#define BATCH 2
#define SEQ 2048
#define TOKENS (BATCH * SEQ) // 4096

typedef _Float16 v8h  __attribute__((ext_vector_type(8)));
typedef _Float16 v16h __attribute__((ext_vector_type(16)));
typedef float    v8f  __attribute__((ext_vector_type(8)));

union V16U { v16h v; v8h h[2]; };

// A-matrix fragment (16x32 f16): lane (m, hi) holds K = {hi*8..hi*8+7, 16+hi*8..16+hi*8+7}
__device__ __forceinline__ v16h ld_a_frag(const _Float16* kbase, int hi) {
  V16U u;
  u.h[0] = *(const v8h*)(kbase + hi * 8);
  u.h[1] = *(const v8h*)(kbase + 16 + hi * 8);
  return u.v;
}
// B-matrix fragment (32x16 f16): lane (n, hi) holds K = {hi*8..hi*8+7, 16+hi*8..16+hi*8+7} of column n
__device__ __forceinline__ v16h ld_b_frag(const _Float16* kbase, int hi) {
  V16U u;
  u.h[0] = *(const v8h*)(kbase + hi * 8);
  u.h[1] = *(const v8h*)(kbase + 16 + hi * 8);
  return u.v;
}
__device__ __forceinline__ v8f wmma_f16(v16h a, v16h b, v8f c) {
  return __builtin_amdgcn_wmma_f32_16x16x32_f16(false, a, false, b, (short)0, c,
                                                false, false);
}

// ---------------- f32 -> f16 conversion ----------------
__global__ void cvt_f32_f16(const float* __restrict__ src, _Float16* __restrict__ dst, int n) {
  int i = blockIdx.x * blockDim.x + threadIdx.x;
  if (i < n) dst[i] = (_Float16)src[i];
}

// ---------------- projections: y = x @ W^T + b ----------------
// grid: (TOKENS/16, D_MODEL/64, 4 matrices), block: 32 (one wave)
__global__ void proj_kernel(const _Float16* __restrict__ xh,
                            const _Float16* __restrict__ wB, const _Float16* __restrict__ wQ,
                            const _Float16* __restrict__ wK, const _Float16* __restrict__ wV,
                            const float* __restrict__ bB, const float* __restrict__ bQ,
                            const float* __restrict__ bK, const float* __restrict__ bV,
                            const float* __restrict__ radial,
                            float* __restrict__ boundary, _Float16* __restrict__ qh,
                            _Float16* __restrict__ kh, _Float16* __restrict__ vth) {
  const int lane = threadIdx.x & 31;
  const int n = lane & 15, hi = lane >> 4;
  const int mt = blockIdx.x;
  const int nb = blockIdx.y;
  const int mat = blockIdx.z;

  const _Float16* W = (mat == 0) ? wB : (mat == 1) ? wQ : (mat == 2) ? wK : wV;
  const float* bias = (mat == 0) ? bB : (mat == 1) ? bQ : (mat == 2) ? bK : bV;

  const _Float16* arow = xh + (size_t)(mt * 16 + n) * D_MODEL;

  v8f acc[4] = {{}, {}, {}, {}};
  for (int kk = 0; kk < D_MODEL / 32; ++kk) {
    v16h a = ld_a_frag(arow + kk * 32, hi);
#pragma unroll
    for (int nt = 0; nt < 4; ++nt) {
      // W row-major [N, K]: column n of B-operand == row (nb*64+nt*16+n) of W
      const _Float16* wcol = W + (size_t)(nb * 64 + nt * 16 + n) * D_MODEL + kk * 32;
      v16h bf = ld_b_frag(wcol, hi);
      acc[nt] = wmma_f16(a, bf, acc[nt]);
    }
  }

#pragma unroll
  for (int nt = 0; nt < 4; ++nt) {
#pragma unroll
    for (int r = 0; r < 8; ++r) {
      int m = mt * 16 + r + hi * 8;          // token index
      int col = nb * 64 + nt * 16 + n;       // model dim
      float val = acc[nt][r] + bias[col];
      int bb_ = m >> 11, s = m & (SEQ - 1);
      int hh = col >> 6, d = col & 63;
      if (mat == 0) {
        boundary[(size_t)m * D_MODEL + col] = val;
      } else if (mat == 1) {
        float sg = 1.f / (1.f + __expf(-radial[hh]));       // sigmoid(radial_weights[h])
        val = val * sg * 0.125f;                            // fold 1/sqrt(64)
        qh[(((size_t)(bb_ * N_HEADS + hh)) * SEQ + s) * HEAD_DIM + d] = (_Float16)val;
      } else if (mat == 2) {
        kh[(((size_t)(bb_ * N_HEADS + hh)) * SEQ + s) * HEAD_DIM + d] = (_Float16)val;
      } else { // V stored transposed [B,H,hd,S] for contiguous B-operand columns
        vth[(((size_t)(bb_ * N_HEADS + hh)) * HEAD_DIM + d) * SEQ + s] = (_Float16)val;
      }
    }
  }
}

// ---------------- attention ----------------
#define AWAVES 2
#define ESTRIDE (SEQ + 8) // pad LDS rows to avoid bank conflicts on strided A loads

__global__ void attn_kernel(const _Float16* __restrict__ qh, const _Float16* __restrict__ kh,
                            const _Float16* __restrict__ vth, const float* __restrict__ boundary,
                            const float* __restrict__ tscale_p,
                            float* __restrict__ out, float* __restrict__ attn) {
  extern __shared__ char smem[];
  const int lane = threadIdx.x & 31;
  const int wave = threadIdx.x >> 5;
  const int n = lane & 15;
  const int hi = lane >> 4;

  _Float16* e_w = (_Float16*)smem + (size_t)wave * 16 * ESTRIDE;
  float* inv_w = (float*)(smem + (size_t)AWAVES * 16 * ESTRIDE * 2) + wave * 16;

  int qt_global = blockIdx.x * AWAVES + wave;   // 0..4095
  int bh = qt_global >> 7;                      // / (SEQ/16)
  int q0 = (qt_global & 127) * 16;
  int b = bh >> 4, h = bh & 15;

  const _Float16* qbh = qh + (size_t)bh * SEQ * HEAD_DIM;
  const _Float16* kbh = kh + (size_t)bh * SEQ * HEAD_DIM;
  const _Float16* vbh = vth + (size_t)bh * HEAD_DIM * SEQ;
  float ts = tscale_p[0];

  // Q A-fragments for the two K=32 steps covering head_dim=64 (load once)
  const _Float16* qrow = qbh + (size_t)(q0 + n) * HEAD_DIM;
  v16h a0 = ld_a_frag(qrow, hi);
  v16h a1 = ld_a_frag(qrow + 32, hi);

  // ---- pass 1: scores via WMMA, running per-row max ----
  float rmax[8];
#pragma unroll
  for (int r = 0; r < 8; ++r) rmax[r] = -3.0e38f;
  for (int kt = 0; kt < SEQ / 16; ++kt) {
    const _Float16* kcol = kbh + (size_t)(kt * 16 + n) * HEAD_DIM;
    v16h b0 = ld_b_frag(kcol, hi);
    v16h b1 = ld_b_frag(kcol + 32, hi);
    v8f c = {};
    c = wmma_f16(a0, b0, c);
    c = wmma_f16(a1, b1, c);
#pragma unroll
    for (int r = 0; r < 8; ++r) rmax[r] = fmaxf(rmax[r], c[r]);
  }
#pragma unroll
  for (int r = 0; r < 8; ++r) {
    float v = rmax[r];
    v = fmaxf(v, __shfl_xor(v, 1, 32));
    v = fmaxf(v, __shfl_xor(v, 2, 32));
    v = fmaxf(v, __shfl_xor(v, 4, 32));
    v = fmaxf(v, __shfl_xor(v, 8, 32));
    rmax[r] = v;
  }

  // ---- pass 2: recompute scores (cheap), e=exp((s-max)*scale) -> LDS, row sums ----
  float rsum[8];
#pragma unroll
  for (int r = 0; r < 8; ++r) rsum[r] = 0.f;
  for (int kt = 0; kt < SEQ / 16; ++kt) {
    const _Float16* kcol = kbh + (size_t)(kt * 16 + n) * HEAD_DIM;
    v16h b0 = ld_b_frag(kcol, hi);
    v16h b1 = ld_b_frag(kcol + 32, hi);
    v8f c = {};
    c = wmma_f16(a0, b0, c);
    c = wmma_f16(a1, b1, c);
#pragma unroll
    for (int r = 0; r < 8; ++r) {
      float e = __expf((c[r] - rmax[r]) * ts);
      rsum[r] += e;
      e_w[(r + hi * 8) * ESTRIDE + kt * 16 + n] = (_Float16)e;
    }
  }
#pragma unroll
  for (int r = 0; r < 8; ++r) {
    float v = rsum[r];
    v += __shfl_xor(v, 1, 32);
    v += __shfl_xor(v, 2, 32);
    v += __shfl_xor(v, 4, 32);
    v += __shfl_xor(v, 8, 32);
    rsum[r] = v;
  }
  if (n == 0) {
#pragma unroll
    for (int r = 0; r < 8; ++r) inv_w[hi * 8 + r] = 1.f / (rsum[r] + 1e-8f);
  }
  __syncthreads();

  // ---- pass 3a: stream normalized attn (f32) to HBM ----
  float* attn_base = attn + ((size_t)bh * SEQ + q0) * SEQ;
  for (int row = 0; row < 16; ++row) {
    float invr = inv_w[row];
    float* arow = attn_base + (size_t)row * SEQ;
    const _Float16* erow = e_w + row * ESTRIDE;
    for (int i = lane; i < SEQ; i += 32) arow[i] = (float)erow[i] * invr;
  }

  // ---- pass 3b: O = E @ V via WMMA (A from LDS, B from transposed V), fold 1/sum ----
  v8f acc[4] = {{}, {}, {}, {}};
  for (int kk = 0; kk < SEQ / 32; ++kk) {
    const _Float16* abase = e_w + n * ESTRIDE + kk * 32;
    v16h a = ld_a_frag(abase, hi);
#pragma unroll
    for (int nt = 0; nt < 4; ++nt) {
      const _Float16* vcol = vbh + (size_t)(nt * 16 + n) * SEQ + kk * 32;
      v16h bf = ld_b_frag(vcol, hi);
      acc[nt] = wmma_f16(a, bf, acc[nt]);
    }
  }
#pragma unroll
  for (int nt = 0; nt < 4; ++nt) {
#pragma unroll
    for (int r = 0; r < 8; ++r) {
      int row = r + hi * 8;
      int srow = q0 + row;
      int col = h * HEAD_DIM + nt * 16 + n;
      size_t oidx = ((size_t)b * SEQ + srow) * D_MODEL + col;
      out[oidx] = acc[nt][r] * inv_w[row] + boundary[oidx];
    }
  }
}

// ---------------- workspace layout (bytes) ----------------
static constexpr size_t XH_OFF = 0;                         // 4096*1024*2   = 8 MiB
static constexpr size_t WH_OFF = XH_OFF + (size_t)TOKENS * D_MODEL * 2;
static constexpr size_t WMAT_BYTES = (size_t)D_MODEL * D_MODEL * 2;  // 2 MiB each
static constexpr size_t QH_OFF = WH_OFF + 4 * WMAT_BYTES;
static constexpr size_t HEADS_BYTES = (size_t)BATCH * N_HEADS * SEQ * HEAD_DIM * 2; // 8 MiB
static constexpr size_t KH_OFF = QH_OFF + HEADS_BYTES;
static constexpr size_t VT_OFF = KH_OFF + HEADS_BYTES;
static constexpr size_t BD_OFF = VT_OFF + HEADS_BYTES;      // boundary f32: 16 MiB

extern "C" void kernel_launch(void* const* d_in, const int* in_sizes, int n_in,
                              void* d_out, int out_size, void* d_ws, size_t ws_size,
                              hipStream_t stream) {
  const float* x      = (const float*)d_in[0];
  const float* Wb     = (const float*)d_in[1];
  const float* bb     = (const float*)d_in[2];
  const float* Wq     = (const float*)d_in[3];
  const float* bq     = (const float*)d_in[4];
  const float* Wk     = (const float*)d_in[5];
  const float* bk     = (const float*)d_in[6];
  const float* Wv     = (const float*)d_in[7];
  const float* bv     = (const float*)d_in[8];
  const float* radial = (const float*)d_in[9];
  const float* tscale = (const float*)d_in[10];

  char* ws = (char*)d_ws;
  _Float16* xh  = (_Float16*)(ws + XH_OFF);
  _Float16* whB = (_Float16*)(ws + WH_OFF + 0 * WMAT_BYTES);
  _Float16* whQ = (_Float16*)(ws + WH_OFF + 1 * WMAT_BYTES);
  _Float16* whK = (_Float16*)(ws + WH_OFF + 2 * WMAT_BYTES);
  _Float16* whV = (_Float16*)(ws + WH_OFF + 3 * WMAT_BYTES);
  _Float16* qh  = (_Float16*)(ws + QH_OFF);
  _Float16* kh  = (_Float16*)(ws + KH_OFF);
  _Float16* vth = (_Float16*)(ws + VT_OFF);
  float* boundary = (float*)(ws + BD_OFF);

  float* out  = (float*)d_out;
  float* attn = (float*)d_out + (size_t)BATCH * SEQ * D_MODEL;

  // 1) convert activations + weights to f16
  int nx = TOKENS * D_MODEL;
  cvt_f32_f16<<<(nx + 255) / 256, 256, 0, stream>>>(x, xh, nx);
  int nw = D_MODEL * D_MODEL;
  cvt_f32_f16<<<(nw + 255) / 256, 256, 0, stream>>>(Wb, whB, nw);
  cvt_f32_f16<<<(nw + 255) / 256, 256, 0, stream>>>(Wq, whQ, nw);
  cvt_f32_f16<<<(nw + 255) / 256, 256, 0, stream>>>(Wk, whK, nw);
  cvt_f32_f16<<<(nw + 255) / 256, 256, 0, stream>>>(Wv, whV, nw);

  // 2) four projections via WMMA
  proj_kernel<<<dim3(TOKENS / 16, D_MODEL / 64, 4), 32, 0, stream>>>(
      xh, whB, whQ, whK, whV, bb, bq, bk, bv, radial, boundary, qh, kh, vth);

  // 3) attention (one wave per 16-row q-tile, AWAVES tiles per block)
  int total_qtiles = BATCH * N_HEADS * (SEQ / 16); // 4096
  size_t shbytes = (size_t)AWAVES * 16 * ESTRIDE * 2 + (size_t)AWAVES * 16 * 4;
  attn_kernel<<<total_qtiles / AWAVES, 32 * AWAVES, shbytes, stream>>>(
      qh, kh, vth, boundary, tscale, out, attn);
}
